// AudioCrossAttentionLayer_19335942766914
// MI455X (gfx1250) — matrix operand order repaired and checked by reading, not compiled, not run on hardware
//
#include <hip/hip_runtime.h>
#include <hip/hip_bf16.h>

// ---------------------------------------------------------------------------
// AudioCrossAttentionLayer for MI455X (gfx1250, wave32, WMMA).
// Pipeline:
//   f32->f16 converts -> audio proj GEMM -> q/k/v GEMMs (f32 out for LoRA)
//   -> LoRA coef + apply (f16 out; V transposed per-head)
//   -> flash attention (WMMA QK^T and PV, online softmax)
//   -> o-projection GEMM (f32 out to d_out)
// All contractions use v_wmma_f32_16x16x32_f16 (f16 in, f32 accumulate).
// GEMM: 32x64 wave tile (8 WMMA per 6 fragment loads), 128x128 block tile.
// Attention: 8 waves/block (128 queries) to halve K/V re-reads from L2.
// ---------------------------------------------------------------------------

typedef __attribute__((ext_vector_type(16))) _Float16 v16h;
typedef __attribute__((ext_vector_type(8)))  _Float16 v8h;
typedef __attribute__((ext_vector_type(8)))  float    v8f;

#define DIMC     1024
#define HEADS    16
#define HEAD_DIM 64
#define RANK     4
#define AUD      768
#define NQ       4096
#define MKV      1024

#define WMMA_F16(a, b, c) \
  __builtin_amdgcn_wmma_f32_16x16x32_f16(false, (a), false, (b), (short)0, (c), false, false)

// ---- fragment loaders (contiguous k order k = 16*half + s, same in A and B) ----------
// A (16x32 f16): lane = M row (lane&15); lanes<16 hold K 0-15,
// lanes>=16 hold K 16-31 (matches B)  -> kA = (lane>=16)*8, offset 2*kA.
__device__ __forceinline__ v16h ld_frag_a(const _Float16* __restrict__ base, int kA) {
  v8h lo = *(const v8h*)(base + 2 * kA);
  v8h hi = *(const v8h*)(base + 2 * kA + 8);
  v16h r;
#pragma unroll
  for (int i = 0; i < 8; ++i) { r[i] = lo[i]; r[i + 8] = hi[i]; }
  return r;
}
// B (32x16 f16): lane = N col (lane&15); lanes<16 hold K 0-15, lanes>=16 hold
// K 16-31 -> caller adds kB = (lane>=16)*16; 16 contiguous halves.
__device__ __forceinline__ v16h ld_frag_b(const _Float16* __restrict__ base) {
  return *(const v16h*)base;  // 32B, lowers to a b128 pair
}

// ---- elementwise f32 -> f16 ------------------------------------------------
__global__ void f32_to_f16(const float* __restrict__ s, _Float16* __restrict__ d, int n) {
  size_t stride = (size_t)gridDim.x * blockDim.x;
  for (size_t i = (size_t)blockIdx.x * blockDim.x + threadIdx.x; i < (size_t)n; i += stride)
    d[i] = (_Float16)s[i];
}

// ---- NT GEMM: C[M,N] f32 = A[M,K]f16 @ W[N,K]f16^T + bias[N] ---------------
// block = 256 threads = 8 waves (4 along M x 2 along N); wave tile 32x64
// (2x4 WMMA accumulators); block tile 128x128.
__global__ void __launch_bounds__(256)
gemm_nt_f16_f32(const _Float16* __restrict__ A, const _Float16* __restrict__ W,
                const float* __restrict__ bias, float* __restrict__ C,
                int Ndim, int K) {
  const int lane = threadIdx.x & 31;
  const int wv   = threadIdx.x >> 5;
  const int l15  = lane & 15;
  const int half = lane >> 4;
  const int m0 = blockIdx.x * 128 + (wv & 3) * 32;
  const int n0 = blockIdx.y * 128 + (wv >> 2) * 64;
  const int kA = half * 8;
  const int kB = half * 16;

  const _Float16* arow0 = A + (size_t)(m0 + l15) * K;
  const _Float16* arow1 = arow0 + (size_t)16 * K;
  const _Float16* brow  = W + (size_t)(n0 + l15) * K + kB;

  v8f acc[2][4] = {};
  for (int k0 = 0; k0 < K; k0 += 32) {
    v16h a0 = ld_frag_a(arow0 + k0, kA);
    v16h a1 = ld_frag_a(arow1 + k0, kA);
    __builtin_prefetch(arow0 + k0 + 64, 0, 0);          // speculative, GL2
    __builtin_prefetch(brow + k0 + 64, 0, 0);
#pragma unroll
    for (int j = 0; j < 4; ++j) {
      v16h bj = ld_frag_b(brow + (size_t)(j * 16) * K + k0);
      acc[0][j] = WMMA_F16(a0, bj, acc[0][j]);
      acc[1][j] = WMMA_F16(a1, bj, acc[1][j]);
    }
  }
#pragma unroll
  for (int i = 0; i < 2; ++i)
#pragma unroll
    for (int j = 0; j < 4; ++j) {
      const int col = n0 + j * 16 + l15;
      const float bv = bias[col];
      const int rbase = m0 + i * 16 + half * 8;   // C layout: vgpr r -> row rbase+r
#pragma unroll
      for (int r = 0; r < 8; ++r)
        C[(size_t)(rbase + r) * Ndim + col] = acc[i][j][r] + bv;
    }
}

// ---- LoRA: coef[row][r] = dot(t[row,:], A[r,:]) (one wave per row) ---------
__global__ void lora_coef(const float* __restrict__ t, const float* __restrict__ Am,
                          float* __restrict__ coef, int rows, int cols) {
  const int gwarp = (int)((blockIdx.x * blockDim.x + threadIdx.x) >> 5);
  const int lane  = threadIdx.x & 31;
  if (gwarp >= rows) return;
  const float* trow = t + (size_t)gwarp * cols;
  float c0 = 0.f, c1 = 0.f, c2 = 0.f, c3 = 0.f;
  for (int j = lane; j < cols; j += 32) {
    const float xv = trow[j];
    c0 += xv * Am[j];
    c1 += xv * Am[cols + j];
    c2 += xv * Am[2 * cols + j];
    c3 += xv * Am[3 * cols + j];
  }
#pragma unroll
  for (int mk = 1; mk < 32; mk <<= 1) {
    c0 += __shfl_xor(c0, mk, 32); c1 += __shfl_xor(c1, mk, 32);
    c2 += __shfl_xor(c2, mk, 32); c3 += __shfl_xor(c3, mk, 32);
  }
  if (lane == 0) {
    float* o = coef + (size_t)gwarp * RANK;
    o[0] = c0; o[1] = c1; o[2] = c2; o[3] = c3;
  }
}

// ---- LoRA apply: dst = f16(t + (coef @ B^T)/RANK); optional transpose ------
__global__ void lora_apply(const float* __restrict__ t, const float* __restrict__ coef,
                           const float* __restrict__ Bm, _Float16* __restrict__ dst,
                           int rows, int cols, int transpose) {
  const size_t total = (size_t)rows * cols;
  const size_t stride = (size_t)gridDim.x * blockDim.x;
  for (size_t idx = (size_t)blockIdx.x * blockDim.x + threadIdx.x; idx < total; idx += stride) {
    const int i = (int)(idx / cols), j = (int)(idx % cols);
    const float* c = coef + (size_t)i * RANK;
    const float* b = Bm + (size_t)j * RANK;
    const float v = t[idx] + 0.25f * (c[0]*b[0] + c[1]*b[1] + c[2]*b[2] + c[3]*b[3]);
    dst[transpose ? ((size_t)j * rows + i) : idx] = (_Float16)v;
  }
}

// ---- flash attention: Q[NQ,C]f16, K[MKV,C]f16, Vt[C,MKV]f16 -> O[NQ,C]f16 --
// block = 256 threads = 8 waves; wave = 16 queries of one head; online softmax
// over MKV in 32-key blocks. P re-shaped via per-wave LDS into an A-fragment.
// 128 queries per block -> each K/V head slice read once per 32 blocks.
__global__ void __launch_bounds__(256)
flash_attn(const _Float16* __restrict__ Q, const _Float16* __restrict__ Kh,
           const _Float16* __restrict__ Vt, _Float16* __restrict__ O) {
  __shared__ __align__(32) _Float16 lds_p[8][16 * 32];
  const int lane = threadIdx.x & 31;
  const int wv   = threadIdx.x >> 5;
  const int l15  = lane & 15;
  const int half = lane >> 4;
  const int h     = blockIdx.y;
  const int qbase = blockIdx.x * 128 + wv * 16;
  const int kA = half * 8;
  const int kB = half * 16;
  const float scale = 0.125f;  // 1/sqrt(64)

  // Q fragments: 16 queries x d=64 -> two 16x32 A-frags, loaded once.
  const _Float16* qrow = Q + (size_t)(qbase + l15) * DIMC + h * HEAD_DIM;
  const v16h aq0 = ld_frag_a(qrow, kA);
  const v16h aq1 = ld_frag_a(qrow + 32, kA);

  v8f acc[4] = {};              // 16 q x 64 dv output accumulator
  float mrun[8], lrun[8];
#pragma unroll
  for (int r = 0; r < 8; ++r) { mrun[r] = -3.0e38f; lrun[r] = 0.f; }

  _Float16* pl = lds_p[wv];

  for (int kb = 0; kb < MKV; kb += 32) {
    // ---- S = Q K^T for 32 keys (two 16x16 tiles), contraction over d=64 ----
    v8f s0 = {}, s1 = {};
    {
      const _Float16* k0p = Kh + (size_t)(kb + l15) * DIMC + h * HEAD_DIM + kB;
      const _Float16* k1p = k0p + (size_t)16 * DIMC;
      s0 = WMMA_F16(aq0, ld_frag_b(k0p),      s0);
      s0 = WMMA_F16(aq1, ld_frag_b(k0p + 32), s0);
      s1 = WMMA_F16(aq0, ld_frag_b(k1p),      s1);
      s1 = WMMA_F16(aq1, ld_frag_b(k1p + 32), s1);
    }
    // ---- online softmax (rows live across 16-lane halves; xor<16 reduces) --
    float alpha[8];
#pragma unroll
    for (int r = 0; r < 8; ++r) { s0[r] *= scale; s1[r] *= scale; }
#pragma unroll
    for (int r = 0; r < 8; ++r) {
      float t = fmaxf(s0[r], s1[r]);
#pragma unroll
      for (int mk = 1; mk < 16; mk <<= 1) t = fmaxf(t, __shfl_xor(t, mk, 32));
      const float mnew = fmaxf(mrun[r], t);
      alpha[r] = __expf(mrun[r] - mnew);
      mrun[r]  = mnew;
      s0[r] = __expf(s0[r] - mnew);
      s1[r] = __expf(s1[r] - mnew);
      float rs = s0[r] + s1[r];
#pragma unroll
      for (int mk = 1; mk < 16; mk <<= 1) rs += __shfl_xor(rs, mk, 32);
      lrun[r] = lrun[r] * alpha[r] + rs;
    }
#pragma unroll
    for (int t4 = 0; t4 < 4; ++t4)
#pragma unroll
      for (int r = 0; r < 8; ++r) acc[t4][r] *= alpha[r];

    // ---- reshape P (C-layout) -> A-fragment via per-wave LDS ---------------
#pragma unroll
    for (int r = 0; r < 8; ++r) {
      const int rowp = half * 8 + r;
      pl[rowp * 32 + l15]      = (_Float16)s0[r];
      pl[rowp * 32 + 16 + l15] = (_Float16)s1[r];
    }
    __syncthreads();
    const v16h ap = ld_frag_a(pl + l15 * 32, kA);
    // ---- O += P @ V (32 keys x 64 dv; Vt is [d][M] so B-frags contiguous) --
#pragma unroll
    for (int t4 = 0; t4 < 4; ++t4) {
      const _Float16* vp = Vt + (size_t)(h * HEAD_DIM + t4 * 16 + l15) * MKV + kb + kB;
      acc[t4] = WMMA_F16(ap, ld_frag_b(vp), acc[t4]);
    }
    __syncthreads();
  }

  // ---- epilogue: divide by l, store f16 ------------------------------------
#pragma unroll
  for (int r = 0; r < 8; ++r) {
    const float inv = lrun[r] > 0.f ? 1.f / lrun[r] : 0.f;
    const int rowq = qbase + half * 8 + r;
#pragma unroll
    for (int t4 = 0; t4 < 4; ++t4)
      O[(size_t)rowq * DIMC + h * HEAD_DIM + t4 * 16 + l15] = (_Float16)(acc[t4][r] * inv);
  }
}

// ---------------------------------------------------------------------------
extern "C" void kernel_launch(void* const* d_in, const int* in_sizes, int n_in,
                              void* d_out, int out_size, void* d_ws, size_t ws_size,
                              hipStream_t stream) {
  (void)in_sizes; (void)n_in; (void)out_size; (void)ws_size;
  const float* x   = (const float*)d_in[0];
  const float* af  = (const float*)d_in[1];
  const float* q_w = (const float*)d_in[2];
  const float* q_b = (const float*)d_in[3];
  const float* k_w = (const float*)d_in[4];
  const float* k_b = (const float*)d_in[5];
  const float* v_w = (const float*)d_in[6];
  const float* v_b = (const float*)d_in[7];
  const float* o_w = (const float*)d_in[8];
  const float* o_b = (const float*)d_in[9];
  const float* a_w = (const float*)d_in[10];
  const float* a_b = (const float*)d_in[11];
  const float* qA  = (const float*)d_in[12];
  const float* qB  = (const float*)d_in[13];
  const float* kAm = (const float*)d_in[14];
  const float* kBm = (const float*)d_in[15];
  const float* vAm = (const float*)d_in[16];
  const float* vBm = (const float*)d_in[17];

  char* ws = (char*)d_ws;
  size_t off = 0;
  auto alloc = [&](size_t bytes) -> void* {
    void* p = ws + off;
    off += (bytes + 255) & ~(size_t)255;
    return p;
  };
  _Float16* xh   = (_Float16*)alloc((size_t)NQ * DIMC * 2);
  _Float16* afh  = (_Float16*)alloc((size_t)MKV * AUD * 2);
  _Float16* awh  = (_Float16*)alloc((size_t)DIMC * AUD * 2);
  _Float16* qwh  = (_Float16*)alloc((size_t)DIMC * DIMC * 2);
  _Float16* kwh  = (_Float16*)alloc((size_t)DIMC * DIMC * 2);
  _Float16* vwh  = (_Float16*)alloc((size_t)DIMC * DIMC * 2);
  _Float16* owh  = (_Float16*)alloc((size_t)DIMC * DIMC * 2);
  float*    audf = (float*)   alloc((size_t)MKV * DIMC * 4);
  _Float16* audh = (_Float16*)alloc((size_t)MKV * DIMC * 2);
  float*    tq   = (float*)   alloc((size_t)NQ * DIMC * 4);
  float*    tk   = (float*)   alloc((size_t)MKV * DIMC * 4);
  float*    tv   = (float*)   alloc((size_t)MKV * DIMC * 4);
  float*    cq   = (float*)   alloc((size_t)NQ * RANK * 4);
  float*    ck   = (float*)   alloc((size_t)MKV * RANK * 4);
  float*    cv   = (float*)   alloc((size_t)MKV * RANK * 4);
  _Float16* qh   = (_Float16*)alloc((size_t)NQ * DIMC * 2);
  _Float16* kh   = (_Float16*)alloc((size_t)MKV * DIMC * 2);
  _Float16* vth  = (_Float16*)alloc((size_t)DIMC * MKV * 2);
  _Float16* atth = (_Float16*)alloc((size_t)NQ * DIMC * 2);

  auto cvt = [&](const float* s, _Float16* d, int n) {
    int blocks = (n + 255) / 256;
    if (blocks > 8192) blocks = 8192;
    f32_to_f16<<<blocks, 256, 0, stream>>>(s, d, n);
  };

  // 1) precision conversion
  cvt(x,   xh,  NQ * DIMC);
  cvt(af,  afh, MKV * AUD);
  cvt(a_w, awh, DIMC * AUD);
  cvt(q_w, qwh, DIMC * DIMC);
  cvt(k_w, kwh, DIMC * DIMC);
  cvt(v_w, vwh, DIMC * DIMC);
  cvt(o_w, owh, DIMC * DIMC);

  // 2) audio projection: audio = af @ a_w^T + a_b  (then to f16)
  gemm_nt_f16_f32<<<dim3(MKV / 128, DIMC / 128), 256, 0, stream>>>(afh, awh, a_b, audf, DIMC, AUD);
  cvt(audf, audh, MKV * DIMC);

  // 3) base q/k/v projections (f32 out, needed exactly for LoRA)
  gemm_nt_f16_f32<<<dim3(NQ  / 128, DIMC / 128), 256, 0, stream>>>(xh,   qwh, q_b, tq, DIMC, DIMC);
  gemm_nt_f16_f32<<<dim3(MKV / 128, DIMC / 128), 256, 0, stream>>>(audh, kwh, k_b, tk, DIMC, DIMC);
  gemm_nt_f16_f32<<<dim3(MKV / 128, DIMC / 128), 256, 0, stream>>>(audh, vwh, v_b, tv, DIMC, DIMC);

  // 4) LoRA corrections; V written per-head-transposed ([d][M]) for PV frags
  lora_coef<<<NQ / 8,  256, 0, stream>>>(tq, qA,  cq, NQ,  DIMC);
  lora_apply<<<2048,   256, 0, stream>>>(tq, cq, qB,  qh,  NQ,  DIMC, 0);
  lora_coef<<<MKV / 8, 256, 0, stream>>>(tk, kAm, ck, MKV, DIMC);
  lora_apply<<<1024,   256, 0, stream>>>(tk, ck, kBm, kh,  MKV, DIMC, 0);
  lora_coef<<<MKV / 8, 256, 0, stream>>>(tv, vAm, cv, MKV, DIMC);
  lora_apply<<<1024,   256, 0, stream>>>(tv, cv, vBm, vth, MKV, DIMC, 1);

  // 5) flash attention
  flash_attn<<<dim3(NQ / 128, HEADS), 256, 0, stream>>>(qh, kh, vth, atth);

  // 6) output projection -> d_out (f32)
  gemm_nt_f16_f32<<<dim3(NQ / 128, DIMC / 128), 256, 0, stream>>>(atth, owh, o_b,
                                                                  (float*)d_out, DIMC, DIMC);
}
